// MoEFFN_42116449304962
// MI455X (gfx1250) — matrix operand order repaired and checked by reading, not compiled, not run on hardware
//
#include <hip/hip_runtime.h>

#define SEQ     1024
#define BATCH   2
#define EMBED   768
#define NEXP    16
#define FFN_H   1536
#define KSH     1
#define KRT     3
#define T_TOK   (SEQ*BATCH)      // 2048
#define NSLOT   4
#define NROWS   (T_TOK*NSLOT)    // 8192 (t,slot) rows; row NROWS = trash row
#define TILE_M  16
#define MAX_TILES (T_TOK/TILE_M) // 128
#define LDSX_STRIDE (EMBED+4)    // f32 path: 772 words -> conflict-free ds_b64
#define XS_BF   (EMBED+8)        // bf16 path: 776 units (1552B, ==16 mod 256B)
#define KCH     512              // fc2 K-chunk staged in LDS
#define HS_BF   (KCH+8)          // 520 units (1040B, ==16 mod 256B)
#define LN_EPS  1e-5f

typedef float  v2f  __attribute__((ext_vector_type(2)));
typedef float  v8f  __attribute__((ext_vector_type(8)));
typedef __bf16 v8bf __attribute__((ext_vector_type(8)));
typedef __bf16 v16bf __attribute__((ext_vector_type(16)));

static __device__ __forceinline__ v16bf cat8(v8bf a, v8bf b) {
    return __builtin_shufflevector(a, b, 0,1,2,3,4,5,6,7,8,9,10,11,12,13,14,15);
}
static __device__ __forceinline__ float gelu_exact(float v) {
    return 0.5f * v * (1.0f + erff(v * 0.7071067811865475f));
}

// ---------------------------------------------------------------- gate ----
__global__ __launch_bounds__(32)
void moe_gate_kernel(const float* __restrict__ x, const float* __restrict__ gw,
                     const float* __restrict__ gb,
                     int* __restrict__ counts, int* __restrict__ tlist,
                     float* __restrict__ slot_w) {
    const int t    = blockIdx.x;
    const int lane = threadIdx.x;
    __shared__ float sc[NEXP];
    if (lane < NEXP) {
        float acc = gb[lane];
        const float* xr = x + (size_t)t * EMBED;
        #pragma unroll 4
        for (int d = 0; d < EMBED; ++d)
            acc = fmaf(xr[d], gw[d * NEXP + lane], acc);
        sc[lane] = acc;
    }
    __syncthreads();
    if (lane == 0) {
        float p[NEXP];
        float m = -3.0e38f;
        for (int e = 0; e < NEXP; ++e) {
            float s = (e < KSH) ? -1e9f : sc[e];
            p[e] = s;
            m = fmaxf(m, s);
        }
        float sum = 0.0f;
        for (int e = 0; e < NEXP; ++e) { p[e] = expf(p[e] - m); sum += p[e]; }
        float inv = 1.0f / sum;
        for (int e = 0; e < NEXP; ++e) p[e] *= inv;

        int   se[NSLOT];
        float swv[NSLOT];
        se[0] = 0; swv[0] = 1.0f;                 // shared expert, weight 1
        for (int s = 1; s <= KRT; ++s) {          // top-3, lowest-index ties
            int bi = 0; float bv = -1.0f;
            for (int e = 0; e < NEXP; ++e)
                if (p[e] > bv) { bv = p[e]; bi = e; }
            se[s] = bi; swv[s] = bv; p[bi] = -1.0f;
        }
        for (int s = 0; s < NSLOT; ++s) {
            int t4 = t * NSLOT + s;
            slot_w[t4] = swv[s];
            int pos = atomicAdd(&counts[se[s]], 1);
            tlist[se[s] * T_TOK + pos] = t4;
        }
    }
}

// ------------------------------------------------ weight repack (bf16x2) ----
// src: [E][K][N] f32  ->  dh/dl: [E][N][K] bf16 (hi / residual-lo), LDS-tiled
__global__ __launch_bounds__(256)
void moe_repack_kernel(const float* __restrict__ w, __bf16* __restrict__ dh,
                       __bf16* __restrict__ dl, int K, int N) {
    const int e  = blockIdx.z;
    const int kb = blockIdx.y * 32;
    const int nb = blockIdx.x * 32;
    __shared__ __bf16 th[32][33], tl[32][33];
    const int tid = threadIdx.x;
    const int tx = tid & 31, ty = tid >> 5;
    const float* src = w + ((size_t)e * K + kb) * N + nb;
    #pragma unroll
    for (int r = ty; r < 32; r += 8) {
        float v = src[(size_t)r * N + tx];
        __bf16 h = (__bf16)v;
        th[r][tx] = h;
        tl[r][tx] = (__bf16)(v - (float)h);
    }
    __syncthreads();
    __bf16* oh = dh + ((size_t)e * N + nb) * K + kb;
    __bf16* ol = dl + ((size_t)e * N + nb) * K + kb;
    #pragma unroll
    for (int r = ty; r < 32; r += 8) {
        oh[(size_t)r * K + tx] = th[tx][r];
        ol[(size_t)r * K + tx] = tl[tx][r];
    }
}

// ---------------------------------------------------------- fc1 (bf16x3) ----
__global__ __launch_bounds__(256)
void moe_fc1_bf16_kernel(const float* __restrict__ x,
                         const __bf16* __restrict__ w1h, const __bf16* __restrict__ w1l,
                         const float* __restrict__ b1,
                         const int* __restrict__ counts, const int* __restrict__ tlist,
                         float* __restrict__ h_ws) {
    const int e    = blockIdx.x >> 7;
    const int tile = blockIdx.x & (MAX_TILES - 1);
    const int cnt  = counts[e];
    if (tile * TILE_M >= cnt) return;

    __shared__ int    s_t4[TILE_M];
    __shared__ __bf16 s_xh[TILE_M * XS_BF];
    __shared__ __bf16 s_xl[TILE_M * XS_BF];
    const int tid = threadIdx.x;
    if (tid < TILE_M) {
        int i = tile * TILE_M + tid;
        s_t4[tid] = (i < cnt) ? tlist[e * T_TOK + i] : NROWS;   // trash row pad
    }
    __syncthreads();
    {   // stage + hi/lo split of the X tile (16 threads per row, coalesced)
        const int r = tid >> 4, li = tid & 15;
        const int t4 = s_t4[r];
        const int t  = (t4 < NROWS) ? (t4 >> 2) : 0;
        const float* xr = x + (size_t)t * EMBED;
        #pragma unroll 4
        for (int j = 0; j < EMBED / 16; ++j) {
            int c = li + j * 16;
            float v = xr[c];
            __bf16 h = (__bf16)v;
            s_xh[r * XS_BF + c] = h;
            s_xl[r * XS_BF + c] = (__bf16)(v - (float)h);
        }
    }
    __syncthreads();

    const int wave = tid >> 5, lane = tid & 31;
    const int half = lane >> 4, l15 = lane & 15;
    const __bf16* ahb = s_xh + l15 * XS_BF + half * 16;  // A: lanes=M, halves split K
    const __bf16* alb = s_xl + l15 * XS_BF + half * 16;

    for (int jt = 0; jt < (FFN_H / 16) / 8; ++jt) {      // 12 column tiles / wave
        const int n0 = (wave + 8 * jt) * 16;
        const __bf16* bh = w1h + ((size_t)(e * FFN_H + n0 + l15)) * EMBED + half * 16;
        const __bf16* bl = w1l + ((size_t)(e * FFN_H + n0 + l15)) * EMBED + half * 16;
        v8f acc = {0.f,0.f,0.f,0.f,0.f,0.f,0.f,0.f};
        #pragma unroll 2
        for (int k0 = 0; k0 < EMBED; k0 += 32) {
            v16bf ah = cat8(*(const v8bf*)(ahb + k0), *(const v8bf*)(ahb + k0 + 8));
            v16bf al = cat8(*(const v8bf*)(alb + k0), *(const v8bf*)(alb + k0 + 8));
            v16bf bhf = *(const v16bf*)(bh + k0);        // B: lane=N, 16 contig K
            v16bf blf = *(const v16bf*)(bl + k0);
            acc = __builtin_amdgcn_wmma_f32_16x16x32_bf16(false, ah, false, bhf,
                                                          (short)0, acc, false, false);
            acc = __builtin_amdgcn_wmma_f32_16x16x32_bf16(false, ah, false, blf,
                                                          (short)0, acc, false, false);
            acc = __builtin_amdgcn_wmma_f32_16x16x32_bf16(false, al, false, bhf,
                                                          (short)0, acc, false, false);
        }
        const float bias = b1[e * FFN_H + n0 + l15];
        #pragma unroll
        for (int r = 0; r < 8; ++r) {                    // D: vgpr r -> M=r+8*half
            int m  = r + half * 8;
            int t4 = s_t4[m];
            h_ws[(size_t)t4 * FFN_H + n0 + l15] = gelu_exact(acc[r] + bias);
        }
    }
}

// ------------------------------------------------------------ fc1 (f32) ----
__global__ __launch_bounds__(256)
void moe_fc1_f32_kernel(const float* __restrict__ x,
                        const float* __restrict__ w1, const float* __restrict__ b1,
                        const int* __restrict__ counts, const int* __restrict__ tlist,
                        float* __restrict__ h_ws) {
    const int e    = blockIdx.x >> 7;
    const int tile = blockIdx.x & (MAX_TILES - 1);
    const int cnt  = counts[e];
    if (tile * TILE_M >= cnt) return;

    __shared__ int   s_t4[TILE_M];
    __shared__ float s_x[TILE_M * LDSX_STRIDE];
    const int tid = threadIdx.x;
    if (tid < TILE_M) {
        int i = tile * TILE_M + tid;
        s_t4[tid] = (i < cnt) ? tlist[e * T_TOK + i] : NROWS;
    }
    __syncthreads();
    {
        const int r = tid >> 4, li = tid & 15;
        const int t4 = s_t4[r];
        const int t  = (t4 < NROWS) ? (t4 >> 2) : 0;
        const float* xr = x + (size_t)t * EMBED;
        #pragma unroll 4
        for (int j = 0; j < EMBED / 16; ++j) {
            int c = li + j * 16;
            s_x[r * LDSX_STRIDE + c] = xr[c];
        }
    }
    __syncthreads();

    const int wave = tid >> 5, lane = tid & 31;
    const int half = lane >> 4, l15 = lane & 15;
    const float* w1e    = w1 + (size_t)e * EMBED * FFN_H;
    const float* a_base = s_x + l15 * LDSX_STRIDE + half * 2;

    for (int jt = 0; jt < (FFN_H / 16) / 8; ++jt) {
        const int n0 = (wave + 8 * jt) * 16;
        v8f acc = {0.f,0.f,0.f,0.f,0.f,0.f,0.f,0.f};
        const float* ap = a_base;
        const float* bp = w1e + (size_t)(half * 2) * FFN_H + n0 + l15;
        #pragma unroll 4
        for (int k0 = 0; k0 < EMBED; k0 += 4) {
            v2f a = *(const v2f*)ap;
            v2f b; b.x = bp[0]; b.y = bp[FFN_H];
            acc = __builtin_amdgcn_wmma_f32_16x16x4_f32(false, a, false, b,
                                                        (short)0, acc, false, false);
            ap += 4;
            bp += 4 * (size_t)FFN_H;
        }
        const float bias = b1[e * FFN_H + n0 + l15];
        #pragma unroll
        for (int r = 0; r < 8; ++r) {
            int m  = r + half * 8;
            int t4 = s_t4[m];
            h_ws[(size_t)t4 * FFN_H + n0 + l15] = gelu_exact(acc[r] + bias);
        }
    }
}

// ------------------------------------------------------------- ln stats ----
__global__ __launch_bounds__(256)
void moe_lnstats_kernel(const float* __restrict__ h_ws, float* __restrict__ mu_rstd) {
    const int row = blockIdx.x;
    const float* hr = h_ws + (size_t)row * FFN_H;
    __shared__ float ss[256], sq[256];
    const int tid = threadIdx.x;
    float s = 0.f, q = 0.f;
    for (int c = tid; c < FFN_H; c += 256) { float v = hr[c]; s += v; q = fmaf(v, v, q); }
    ss[tid] = s; sq[tid] = q;
    __syncthreads();
    for (int off = 128; off > 0; off >>= 1) {            // fixed tree -> deterministic
        if (tid < off) { ss[tid] += ss[tid + off]; sq[tid] += sq[tid + off]; }
        __syncthreads();
    }
    if (tid == 0) {
        float mu  = ss[0] * (1.0f / FFN_H);
        float var = sq[0] * (1.0f / FFN_H) - mu * mu;
        mu_rstd[2 * row]     = mu;
        mu_rstd[2 * row + 1] = rsqrtf(var + LN_EPS);
    }
}

// ---------------------------------------------------------- fc2 (bf16x3) ----
__global__ __launch_bounds__(256)
void moe_fc2_bf16_kernel(const float* __restrict__ h_ws,
                         const float* __restrict__ lnw, const float* __restrict__ lnb,
                         const __bf16* __restrict__ w2h, const __bf16* __restrict__ w2l,
                         const float* __restrict__ b2,
                         const int* __restrict__ counts, const int* __restrict__ tlist,
                         const float* __restrict__ mu_rstd,
                         const float* __restrict__ slot_w,
                         float* __restrict__ y_slots) {
    const int e    = blockIdx.x >> 7;
    const int tile = blockIdx.x & (MAX_TILES - 1);
    const int cnt  = counts[e];
    if (tile * TILE_M >= cnt) return;

    __shared__ int    s_t4[TILE_M];
    __shared__ float  s_w[TILE_M], s_mu[TILE_M], s_rs[TILE_M];
    __shared__ __bf16 s_ah[TILE_M * HS_BF];
    __shared__ __bf16 s_al[TILE_M * HS_BF];
    const int tid = threadIdx.x;
    if (tid < TILE_M) {
        int i  = tile * TILE_M + tid;
        int t4 = (i < cnt) ? tlist[e * T_TOK + i] : NROWS;
        s_t4[tid] = t4;
        s_w[tid]  = slot_w[t4];
        s_mu[tid] = mu_rstd[2 * t4];
        s_rs[tid] = mu_rstd[2 * t4 + 1];
    }
    __syncthreads();

    const int wave = tid >> 5, lane = tid & 31;
    const int half = lane >> 4, l15 = lane & 15;
    const int r  = tid >> 4, li = tid & 15;              // staging role
    v8f acc[6];
    #pragma unroll
    for (int jt = 0; jt < 6; ++jt) acc[jt] = (v8f){0.f,0.f,0.f,0.f,0.f,0.f,0.f,0.f};

    for (int kc = 0; kc < FFN_H; kc += KCH) {            // 3 K-chunks
        {   // stage LN-normalized h chunk, split hi/lo (LN done once per element)
            const int   t4 = s_t4[r];
            const float mu = s_mu[r], rs = s_rs[r];
            const float* hr = h_ws + (size_t)t4 * FFN_H + kc;
            const float* lw = lnw + (size_t)e * FFN_H + kc;
            const float* lb = lnb + (size_t)e * FFN_H + kc;
            #pragma unroll 4
            for (int j = 0; j < KCH / 16; ++j) {
                int c = li + j * 16;
                float v = fmaf((hr[c] - mu) * rs, lw[c], lb[c]);
                __bf16 h = (__bf16)v;
                s_ah[r * HS_BF + c] = h;
                s_al[r * HS_BF + c] = (__bf16)(v - (float)h);
            }
        }
        __syncthreads();
        const __bf16* ahb = s_ah + l15 * HS_BF + half * 16;
        const __bf16* alb = s_al + l15 * HS_BF + half * 16;
        #pragma unroll
        for (int jt = 0; jt < 6; ++jt) {                 // 6 column tiles / wave
            const int n0 = (wave + 8 * jt) * 16;
            const __bf16* bh = w2h + ((size_t)(e * EMBED + n0 + l15)) * FFN_H + kc + half * 16;
            const __bf16* bl = w2l + ((size_t)(e * EMBED + n0 + l15)) * FFN_H + kc + half * 16;
            v8f a = acc[jt];
            #pragma unroll 2
            for (int k0 = 0; k0 < KCH; k0 += 32) {
                v16bf ah = cat8(*(const v8bf*)(ahb + k0), *(const v8bf*)(ahb + k0 + 8));
                v16bf al = cat8(*(const v8bf*)(alb + k0), *(const v8bf*)(alb + k0 + 8));
                v16bf bhf = *(const v16bf*)(bh + k0);
                v16bf blf = *(const v16bf*)(bl + k0);
                a = __builtin_amdgcn_wmma_f32_16x16x32_bf16(false, ah, false, bhf,
                                                            (short)0, a, false, false);
                a = __builtin_amdgcn_wmma_f32_16x16x32_bf16(false, ah, false, blf,
                                                            (short)0, a, false, false);
                a = __builtin_amdgcn_wmma_f32_16x16x32_bf16(false, al, false, bhf,
                                                            (short)0, a, false, false);
            }
            acc[jt] = a;
        }
        __syncthreads();
    }
    #pragma unroll
    for (int jt = 0; jt < 6; ++jt) {
        const int n0 = (wave + 8 * jt) * 16;
        const float bias = b2[e * EMBED + n0 + l15];
        #pragma unroll
        for (int rr = 0; rr < 8; ++rr) {
            int m  = rr + half * 8;
            int t4 = s_t4[m];
            y_slots[(size_t)t4 * EMBED + n0 + l15] = (acc[jt][rr] + bias) * s_w[m];
        }
    }
}

// ------------------------------------------------------------ fc2 (f32) ----
__global__ __launch_bounds__(256)
void moe_fc2_f32_kernel(const float* __restrict__ h_ws,
                        const float* __restrict__ lnw, const float* __restrict__ lnb,
                        const float* __restrict__ w2,  const float* __restrict__ b2,
                        const int* __restrict__ counts, const int* __restrict__ tlist,
                        const float* __restrict__ mu_rstd,
                        const float* __restrict__ slot_w,
                        float* __restrict__ y_slots) {
    const int e    = blockIdx.x >> 7;
    const int tile = blockIdx.x & (MAX_TILES - 1);
    const int cnt  = counts[e];
    if (tile * TILE_M >= cnt) return;

    __shared__ int   s_t4[TILE_M];
    __shared__ float s_w[TILE_M];
    __shared__ float s_lnw[FFN_H];
    __shared__ float s_lnb[FFN_H];
    const int tid = threadIdx.x;
    if (tid < TILE_M) {
        int i  = tile * TILE_M + tid;
        int t4 = (i < cnt) ? tlist[e * T_TOK + i] : NROWS;
        s_t4[tid] = t4;
        s_w[tid]  = slot_w[t4];
    }
    for (int c = tid; c < FFN_H; c += 256) {
        s_lnw[c] = lnw[e * FFN_H + c];
        s_lnb[c] = lnb[e * FFN_H + c];
    }
    __syncthreads();

    const int wave = tid >> 5, lane = tid & 31;
    const int half = lane >> 4, l15 = lane & 15;
    const int   my_t4 = s_t4[l15];
    const float mu    = mu_rstd[2 * my_t4];
    const float rstd  = mu_rstd[2 * my_t4 + 1];
    const float* hrow = h_ws + (size_t)my_t4 * FFN_H + half * 2;
    const float* w2e  = w2 + (size_t)e * FFN_H * EMBED;

    for (int jt = 0; jt < (EMBED / 16) / 8; ++jt) {
        const int n0 = (wave + 8 * jt) * 16;
        v8f acc = {0.f,0.f,0.f,0.f,0.f,0.f,0.f,0.f};
        const float* ap  = hrow;
        const float* bp  = w2e + (size_t)(half * 2) * EMBED + n0 + l15;
        const float* lwp = s_lnw + half * 2;
        const float* lbp = s_lnb + half * 2;
        #pragma unroll 4
        for (int k0 = 0; k0 < FFN_H; k0 += 4) {
            v2f hv = *(const v2f*)ap;
            v2f a;
            a.x = fmaf((hv.x - mu) * rstd, lwp[0], lbp[0]);
            a.y = fmaf((hv.y - mu) * rstd, lwp[1], lbp[1]);
            v2f b; b.x = bp[0]; b.y = bp[EMBED];
            acc = __builtin_amdgcn_wmma_f32_16x16x4_f32(false, a, false, b,
                                                        (short)0, acc, false, false);
            ap  += 4;
            bp  += 4 * (size_t)EMBED;
            lwp += 4; lbp += 4;
        }
        const float bias = b2[e * EMBED + n0 + l15];
        #pragma unroll
        for (int r = 0; r < 8; ++r) {
            int m  = r + half * 8;
            int t4 = s_t4[m];
            y_slots[(size_t)t4 * EMBED + n0 + l15] = (acc[r] + bias) * s_w[m];
        }
    }
}

// -------------------------------------------------------------- combine ----
__global__ __launch_bounds__(256)
void moe_combine_kernel(const float* __restrict__ y_slots, float* __restrict__ out) {
    const int idx = blockIdx.x * 256 + threadIdx.x;
    if (idx >= T_TOK * EMBED) return;
    const int t = idx / EMBED, d = idx - t * EMBED;
    const float* yb = y_slots + (size_t)(t * NSLOT) * EMBED + d;
    out[idx] = ((yb[0] + yb[EMBED]) + yb[2 * EMBED]) + yb[3 * EMBED]; // fixed order
}

// ---------------------------------------------------------------- launch ----
extern "C" void kernel_launch(void* const* d_in, const int* in_sizes, int n_in,
                              void* d_out, int out_size, void* d_ws, size_t ws_size,
                              hipStream_t stream) {
    const float* x      = (const float*)d_in[0];
    const float* gate_w = (const float*)d_in[1];
    const float* gate_b = (const float*)d_in[2];
    const float* fc1_w  = (const float*)d_in[3];
    const float* fc1_b  = (const float*)d_in[4];
    const float* ln_w   = (const float*)d_in[5];
    const float* ln_b   = (const float*)d_in[6];
    const float* fc2_w  = (const float*)d_in[7];
    const float* fc2_b  = (const float*)d_in[8];
    float* out = (float*)d_out;
    (void)in_sizes; (void)n_in; (void)out_size;

    char* ws = (char*)d_ws;
    size_t off = 0;
    auto carve = [&](size_t bytes) { char* p = ws + off; off = (off + bytes + 255) & ~(size_t)255; return p; };
    int*    counts  = (int*)   carve(NEXP * sizeof(int));
    int*    tlist   = (int*)   carve((size_t)NEXP * T_TOK * sizeof(int));
    float*  slot_w  = (float*) carve((size_t)(NROWS + 1) * sizeof(float));
    float*  mu_rstd = (float*) carve((size_t)2 * (NROWS + 1) * sizeof(float));
    float*  h_ws    = (float*) carve((size_t)(NROWS + 1) * FFN_H * sizeof(float));
    float*  y_slots = (float*) carve((size_t)(NROWS + 1) * EMBED * sizeof(float));
    const size_t need_f32 = off;
    __bf16* w1h = (__bf16*)carve((size_t)NEXP * FFN_H * EMBED * sizeof(__bf16));
    __bf16* w1l = (__bf16*)carve((size_t)NEXP * FFN_H * EMBED * sizeof(__bf16));
    __bf16* w2h = (__bf16*)carve((size_t)NEXP * EMBED * FFN_H * sizeof(__bf16));
    __bf16* w2l = (__bf16*)carve((size_t)NEXP * EMBED * FFN_H * sizeof(__bf16));
    const size_t need_bf16 = off;
    const bool use_bf16 = (ws_size >= need_bf16);
    (void)need_f32;

    hipMemsetAsync(counts, 0, NEXP * sizeof(int), stream);   // fresh lists each call

    moe_gate_kernel<<<dim3(T_TOK), dim3(32), 0, stream>>>(
        x, gate_w, gate_b, counts, tlist, slot_w);

    if (use_bf16) {
        // repack weights into transposed bf16 hi/lo (L2-resident)
        moe_repack_kernel<<<dim3(FFN_H / 32, EMBED / 32, NEXP), dim3(256), 0, stream>>>(
            fc1_w, w1h, w1l, EMBED, FFN_H);
        moe_repack_kernel<<<dim3(EMBED / 32, FFN_H / 32, NEXP), dim3(256), 0, stream>>>(
            fc2_w, w2h, w2l, FFN_H, EMBED);
        moe_fc1_bf16_kernel<<<dim3(NEXP * MAX_TILES), dim3(256), 0, stream>>>(
            x, w1h, w1l, fc1_b, counts, tlist, h_ws);
        moe_lnstats_kernel<<<dim3(NROWS), dim3(256), 0, stream>>>(h_ws, mu_rstd);
        moe_fc2_bf16_kernel<<<dim3(NEXP * MAX_TILES), dim3(256), 0, stream>>>(
            h_ws, ln_w, ln_b, w2h, w2l, fc2_b, counts, tlist, mu_rstd, slot_w, y_slots);
    } else {
        moe_fc1_f32_kernel<<<dim3(NEXP * MAX_TILES), dim3(256), 0, stream>>>(
            x, fc1_w, fc1_b, counts, tlist, h_ws);
        moe_lnstats_kernel<<<dim3(NROWS), dim3(256), 0, stream>>>(h_ws, mu_rstd);
        moe_fc2_f32_kernel<<<dim3(NEXP * MAX_TILES), dim3(256), 0, stream>>>(
            h_ws, ln_w, ln_b, fc2_w, fc2_b, counts, tlist, mu_rstd, slot_w, y_slots);
    }

    const int nelem = T_TOK * EMBED;
    moe_combine_kernel<<<dim3((nelem + 255) / 256), dim3(256), 0, stream>>>(y_slots, out);
}
